// SeparableConv2d_83708912599746
// MI455X (gfx1250) — matrix operand order repaired and checked by reading, not compiled, not run on hardware
//
#include <hip/hip_runtime.h>

// MI455X (gfx1250) fused NaN-aware depthwise(3x3) + pointwise(1x1) conv.
//
// Roofline: fused traffic = 128MB (x in) + 256MB (out) ~= 385MB => ~16.5us
// at 23.3 TB/s. Pointwise GEMM = 17.2 GFLOP; f32 WMMA (16x16x4) would make
// this compute bound, so we use the bf16 split trick (hi+lo, 3 products,
// ~fp32 accuracy) on v_wmma_f32_16x16x32_bf16 to stay on the memory roofline.

typedef __attribute__((ext_vector_type(16))) __bf16 v16bf;
typedef __attribute__((ext_vector_type(8)))  float  v8f;

#define TILE_P 64          // pixels per workgroup (one row segment)
#define C_IN   128
#define C_OUT  256
#define HDIM   256
#define WDIM   256
#define HW     (HDIM * WDIM)

__device__ __forceinline__ float qnanf() {
    union { unsigned u; float f; } u; u.u = 0x7FC00000u; return u.f;
}

__global__ __launch_bounds__(256, 1)
void sepconv_nan_fused(const float* __restrict__ x,
                       const float* __restrict__ wdw,
                       const float* __restrict__ wpw,
                       float* __restrict__ out) {
    // Pointwise weights, pre-permuted into WMMA A-matrix per-lane order, bf16 hi/lo.
    // Fragment (mt, k0, lane) -> 16 contiguous bf16 (one 32B ds load per lane).
    __shared__ __align__(32) __bf16 sAhi[16 * 4 * 32 * 16];   // 64 KB
    __shared__ __align__(32) __bf16 sAlo[16 * 4 * 32 * 16];   // 64 KB
    // Depthwise output tile, K(channel)-contiguous per pixel = B-matrix friendly.
    __shared__ __align__(32) __bf16 sYhi[TILE_P * C_IN];      // 16 KB
    __shared__ __align__(32) __bf16 sYlo[TILE_P * C_IN];      // 16 KB
    __shared__ float sWdw[C_IN * 9];
    __shared__ int   sPnan[TILE_P];                           // NaN channels per pixel

    const int t    = threadIdx.x;
    const int tile = blockIdx.x;            // 4096 tiles total
    const int b    = tile >> 10;            // batch
    const int rem  = tile & 1023;
    const int h    = rem >> 2;              // row
    const int w0   = (rem & 3) * TILE_P;    // row segment start

    // ---- Stage 0: weights to LDS ------------------------------------------
    for (int i = t; i < C_IN * 9; i += 256) sWdw[i] = wdw[i];
    if (t < TILE_P) sPnan[t] = 0;

    // Permute w_pw[256,128] into per-lane A fragments (16x32 tile), this code's contiguous k order:
    // lanes 0-15 hold row M=lane, elems j -> K=j ; lanes 16-31: K+=16.
    for (int idx = t; idx < 16 * 4 * 32; idx += 256) {
        const int lane = idx & 31;
        const int k0   = (idx >> 5) & 3;
        const int mt   = idx >> 7;
        const int m    = mt * 16 + (lane & 15);
        const int kb   = k0 * 32 + ((lane & 16) ? 16 : 0);
        #pragma unroll
        for (int j = 0; j < 16; ++j) {
            const int k = kb + j;
            const float v = wpw[m * C_IN + k];
            const __bf16 hi = (__bf16)v;
            const __bf16 lo = (__bf16)(v - (float)hi);
            sAhi[idx * 16 + j] = hi;
            sAlo[idx * 16 + j] = lo;
        }
    }
    __syncthreads();

    // ---- Stage 1: NaN-aware depthwise 3x3 into LDS (bf16 hi/lo) ----------
    {
        const int p  = t & 63;        // pixel within tile
        const int cb = t >> 6;        // channel block (32 channels each)
        const int w  = w0 + p;
        int localNanChans = 0;
        for (int ci = 0; ci < 32; ++ci) {
            const int c = cb * 32 + ci;
            const float* xp = x + (size_t)(b * C_IN + c) * HW;
            float acc = 0.0f;
            int   cnt = 0;
            #pragma unroll
            for (int dh = -1; dh <= 1; ++dh) {
                const int hh = h + dh;
                if ((unsigned)hh < (unsigned)HDIM) {
                    const float* rowp = xp + hh * WDIM;
                    #pragma unroll
                    for (int dw = -1; dw <= 1; ++dw) {
                        const int ww = w + dw;
                        if ((unsigned)ww < (unsigned)WDIM) {
                            const float v = rowp[ww];
                            const bool nan = (v != v);
                            cnt += nan ? 1 : 0;
                            const float wv = sWdw[c * 9 + (dh + 1) * 3 + (dw + 1)];
                            acc += nan ? 0.0f : v * wv;
                        }
                    }
                }
            }
            __bf16 hi, lo;
            if (cnt > 4) {                    // frac > 0.5 of 9 -> y is NaN
                ++localNanChans;
                hi = (__bf16)0.0f; lo = (__bf16)0.0f;   // NaN contributes 0 downstream
            } else {
                hi = (__bf16)acc;
                lo = (__bf16)(acc - (float)hi);
            }
            sYhi[p * C_IN + c] = hi;
            sYlo[p * C_IN + c] = lo;
        }
        if (localNanChans) atomicAdd(&sPnan[p], localNanChans);
    }
    __syncthreads();

    // ---- Stage 2: pointwise GEMM via bf16 WMMA (hi*hi + hi*lo + lo*hi) ---
    const int wave = t >> 5;
    const int lane = t & 31;

    v8f acc[2][4];
    #pragma unroll
    for (int i = 0; i < 2; ++i)
        #pragma unroll
        for (int nt = 0; nt < 4; ++nt)
            acc[i][nt] = (v8f){0.f, 0.f, 0.f, 0.f, 0.f, 0.f, 0.f, 0.f};

    for (int k0 = 0; k0 < 4; ++k0) {        // K = 128 in steps of 32
        v16bf ahi[2], alo[2];
        #pragma unroll
        for (int i = 0; i < 2; ++i) {
            const int mt  = wave * 2 + i;
            const int off = ((mt * 4 + k0) * 32 + lane) * 16;
            ahi[i] = *(const v16bf*)&sAhi[off];
            alo[i] = *(const v16bf*)&sAlo[off];
        }
        #pragma unroll
        for (int nt = 0; nt < 4; ++nt) {
            // B per-lane, same contiguous k order as A: col n = lane&15; lanes 0-15 K=0..15, 16-31 K=16..31
            const int p    = nt * 16 + (lane & 15);
            const int koff = k0 * 32 + ((lane & 16) ? 16 : 0);
            const v16bf bhi = *(const v16bf*)&sYhi[p * C_IN + koff];
            const v16bf blo = *(const v16bf*)&sYlo[p * C_IN + koff];
            #pragma unroll
            for (int i = 0; i < 2; ++i) {
                acc[i][nt] = __builtin_amdgcn_wmma_f32_16x16x32_bf16(
                    false, ahi[i], false, bhi, (short)0, acc[i][nt], false, false);
                acc[i][nt] = __builtin_amdgcn_wmma_f32_16x16x32_bf16(
                    false, ahi[i], false, blo, (short)0, acc[i][nt], false, false);
                acc[i][nt] = __builtin_amdgcn_wmma_f32_16x16x32_bf16(
                    false, alo[i], false, bhi, (short)0, acc[i][nt], false, false);
            }
        }
    }

    // ---- Epilogue: per-pixel channel-NaN gate, scatter to NCHW -----------
    // C/D layout: VGPR r, lane l -> M = r + 8*(l>=16), N = l&15.
    const float NANF = qnanf();
    #pragma unroll
    for (int i = 0; i < 2; ++i) {
        const int mt = wave * 2 + i;
        const int coutBase = mt * 16 + ((lane & 16) ? 8 : 0);
        #pragma unroll
        for (int nt = 0; nt < 4; ++nt) {
            const int n = nt * 16 + (lane & 15);
            const bool pixNan = sPnan[n] > 64;      // frac > 0.5 of 128 channels
            const int wcol = w0 + n;
            #pragma unroll
            for (int r = 0; r < 8; ++r) {
                const int cout = coutBase + r;
                const float v = pixNan ? NANF : acc[i][nt][r];
                out[(size_t)(b * C_OUT + cout) * HW + h * WDIM + wcol] = v;
            }
        }
    }
}

extern "C" void kernel_launch(void* const* d_in, const int* in_sizes, int n_in,
                              void* d_out, int out_size, void* d_ws, size_t ws_size,
                              hipStream_t stream) {
    (void)in_sizes; (void)n_in; (void)out_size; (void)d_ws; (void)ws_size;
    const float* x    = (const float*)d_in[0];   // [4,128,256,256]
    const float* w_dw = (const float*)d_in[1];   // [128,1,3,3]
    const float* w_pw = (const float*)d_in[2];   // [256,128,1,1]
    float* out        = (float*)d_out;           // [4,256,256,256]

    // 4 batches * 256 rows * 4 segments of 64 pixels = 4096 workgroups
    dim3 grid(4096), block(256);
    sepconv_nan_fused<<<grid, block, 0, stream>>>(x, w_dw, w_pw, out);
}
